// MultiLayer_LSTM_14791867368063
// MI455X (gfx1250) — compile-verified, matrix operand order checked by reading, not run on hardware
//
#include <hip/hip_runtime.h>

typedef __attribute__((ext_vector_type(16))) _Float16 v16h;
typedef __attribute__((ext_vector_type(8)))  _Float16 v8h;
typedef __attribute__((ext_vector_type(8)))  float    v8f;

#define BATCH 32
#define TLEN  1024
#define DIM   512      // D == U == 512
#define GDIM  2048     // 4*U
#define NWG   32       // workgroups (u-slices)
#define UW    16       // u columns per WG
#define KTOT  1024     // K = D (x part) + U (h part)
#define KC_N  32       // KTOT / 32
#define HS    (BATCH*DIM)   // elements per h ping-pong buffer

// LDS partition (bytes)
#define LDS_WB_BYTES   (KC_N*4*32*16*2)              // 131072: weight fragments
#define LDS_G_BYTES    (4*32*17*4)                   // 8704:  gate staging (padded)
#define LDS_C_BYTES    (BATCH*UW*4)                  // 2048:  cell state slice
#define LDS_B_BYTES    (64*4)                        // 256:   bias slice
#define LDS_TOTAL      (LDS_WB_BYTES+LDS_G_BYTES+LDS_C_BYTES+LDS_B_BYTES)

__device__ __forceinline__ float fsig(float x)  { return 1.0f / (1.0f + __expf(-x)); }
__device__ __forceinline__ float ftanh(float x) { return 1.0f - 2.0f / (__expf(2.0f * x) + 1.0f); }

__global__ void reset_counter_kernel(int* cnt) {
    if (threadIdx.x == 0) *cnt = 0;
}

__global__ void cast_f32_to_f16_kernel(const float* __restrict__ in,
                                       _Float16* __restrict__ out, long long n) {
    long long i = (long long)blockIdx.x * blockDim.x + threadIdx.x;
    if (i < n) out[i] = (_Float16)in[i];
}

__global__ __launch_bounds__(256, 1)
void lstm_layer_kernel(const _Float16* __restrict__ xin,   // [B,T,512] f16 layer input
                       _Float16* __restrict__ xout,        // [B,T,512] f16 layer output (if !is_last)
                       _Float16* __restrict__ hbuf,        // [2][B,512] f16 ping-pong hidden
                       int* __restrict__ cnt,              // global barrier counter (pre-zeroed)
                       const float* __restrict__ W,        // [512,2048] f32
                       const float* __restrict__ R,        // [512,2048] f32
                       const float* __restrict__ bias,     // [2048] f32
                       float* __restrict__ out_seq,        // [B,T,512] f32 (if is_last)
                       float* __restrict__ out_h,          // [B,512] f32 (this layer)
                       float* __restrict__ out_c,          // [B,512] f32 (this layer)
                       int is_last)
{
    extern __shared__ unsigned char smem[];
    _Float16* wB  = (_Float16*)smem;                                        // [KC_N][4][32][16]
    float*    gb  = (float*)(smem + LDS_WB_BYTES);                          // [4][32][17]
    float*    cb  = (float*)(smem + LDS_WB_BYTES + LDS_G_BYTES);            // [32][16]
    float*    bl  = (float*)(smem + LDS_WB_BYTES + LDS_G_BYTES + LDS_C_BYTES); // [64]

    const int tid   = threadIdx.x;
    const int wg    = blockIdx.x;
    const int u0    = wg * UW;
    const int lane  = tid & 31;
    const int wid   = tid >> 5;     // 0..7
    const int nt    = wid & 3;      // gate tile: 0=i 1=f 2=cand 3=o
    const int mt    = wid >> 2;     // 0..1 batch half
    const int l15   = lane & 15;
    const int lh    = lane >> 4;    // 0/1 lane half
    const int bbrow = mt * 16 + l15;  // batch row this lane feeds into A fragments

    // ---- one-time: stage [W;R] slice into LDS pre-swizzled as WMMA-B fragments ----
    // B fragment (32x16 f16) per (kc,nt): lane n<16 holds col N=n, K = base+{0..7,16..23};
    // lane n+16 holds col N=n, K = base+{8..15,24..31}; 16 f16 contiguous per lane.
    for (int e = tid; e < KC_N * 4 * 32 * 16; e += 256) {
        int j   = e & 15;
        int ln  = (e >> 4) & 31;
        int g   = (e >> 9) & 3;
        int kc  = e >> 11;
        int koff = ((ln >> 4) * 8) + ((j < 8) ? j : (j + 8));
        int K    = kc * 32 + koff;
        int col  = g * DIM + u0 + (ln & 15);
        float v  = (K < DIM) ? W[(size_t)K * GDIM + col]
                             : R[(size_t)(K - DIM) * GDIM + col];
        wB[e] = (_Float16)v;
    }
    if (tid < 64) bl[tid] = bias[(tid >> 4) * DIM + u0 + (tid & 15)];
    for (int e = tid; e < BATCH * UW; e += 256) {
        cb[e] = 0.0f;
        hbuf[(size_t)(e >> 4) * DIM + u0 + (e & 15)] = (_Float16)0.0f;  // parity 0 = h_{-1}
    }

    // ---- global barrier 0: all h(0) slices + weight staging visible ----
    __syncthreads();
    if (tid == 0) {
        __threadfence();
        atomicAdd(cnt, 1);
        volatile int* vc = cnt;
        while (*vc < NWG) __builtin_amdgcn_s_sleep(1);
        __threadfence();
    }
    __syncthreads();

    // ---- serial scan over T ----
    for (int t = 0; t < TLEN; ++t) {
        const _Float16* xrow = xin  + ((size_t)bbrow * TLEN + t) * DIM;
        const _Float16* hrow = hbuf + (size_t)(t & 1) * HS + (size_t)bbrow * DIM;
        if (t + 1 < TLEN)
            __builtin_prefetch(xin + ((size_t)bbrow * TLEN + (t + 1)) * DIM, 0, 3);

        v8f acc = {0.f, 0.f, 0.f, 0.f, 0.f, 0.f, 0.f, 0.f};
        const int lbase = lh * 8;

        // K chunks 0..15: x_t @ W slice
        #pragma unroll 4
        for (int kc = 0; kc < 16; ++kc) {
            const _Float16* src = xrow + kc * 32 + lbase;
            v8h lo = *(const v8h*)(src);
            v8h hi = *(const v8h*)(src + 16);
            v16h a = __builtin_shufflevector(lo, hi, 0,1,2,3,4,5,6,7,8,9,10,11,12,13,14,15);
            v16h bf = *(const v16h*)(wB + (((kc * 4 + nt) * 32 + lane) * 16));
            acc = __builtin_amdgcn_wmma_f32_16x16x32_f16(false, a, false, bf,
                                                         (short)0, acc, false, false);
        }
        // K chunks 16..31: h_{t-1} @ R slice
        #pragma unroll 4
        for (int kc = 16; kc < 32; ++kc) {
            const _Float16* src = hrow + (kc - 16) * 32 + lbase;
            v8h lo = *(const v8h*)(src);
            v8h hi = *(const v8h*)(src + 16);
            v16h a = __builtin_shufflevector(lo, hi, 0,1,2,3,4,5,6,7,8,9,10,11,12,13,14,15);
            v16h bf = *(const v16h*)(wB + (((kc * 4 + nt) * 32 + lane) * 16));
            acc = __builtin_amdgcn_wmma_f32_16x16x32_f16(false, a, false, bf,
                                                         (short)0, acc, false, false);
        }

        // scatter gate tiles to LDS: C/D layout -> VGPR r, lane l: (M = r + 8*lh, N = l15)
        #pragma unroll
        for (int r = 0; r < 8; ++r)
            gb[(nt * 32 + mt * 16 + lh * 8 + r) * 17 + l15] = acc[r];
        __syncthreads();

        // elementwise LSTM cell update for this WG's 32x16 slice
        for (int e = tid; e < BATCH * UW; e += 256) {
            int bb = e >> 4, u = e & 15;
            float gi = gb[(0 * 32 + bb) * 17 + u] + bl[ 0 + u];
            float gf = gb[(1 * 32 + bb) * 17 + u] + bl[16 + u];
            float gc = gb[(2 * 32 + bb) * 17 + u] + bl[32 + u];
            float go = gb[(3 * 32 + bb) * 17 + u] + bl[48 + u];
            float cold = cb[bb * 16 + u];
            float cn = fsig(gf) * cold + fsig(gi) * ftanh(gc);
            float hn = fsig(go) * ftanh(cn);
            cb[bb * 16 + u] = cn;
            hbuf[(size_t)((t + 1) & 1) * HS + (size_t)bb * DIM + u0 + u] = (_Float16)hn;
            if (is_last)
                out_seq[((size_t)bb * TLEN + t) * DIM + u0 + u] = hn;
            else
                xout[((size_t)bb * TLEN + t) * DIM + u0 + u] = (_Float16)hn;
            if (t == TLEN - 1) {
                out_h[bb * DIM + u0 + u] = hn;
                out_c[bb * DIM + u0 + u] = cn;
            }
        }

        // ---- per-step device barrier (release h_t, reuse gbuf safely) ----
        __syncthreads();
        if (tid == 0) {
            __threadfence();
            atomicAdd(cnt, 1);
            const int target = NWG * (t + 2);
            volatile int* vc = cnt;
            while (*vc < target) __builtin_amdgcn_s_sleep(1);
            __threadfence();
        }
        __syncthreads();
    }
}

extern "C" void kernel_launch(void* const* d_in, const int* in_sizes, int n_in,
                              void* d_out, int out_size, void* d_ws, size_t ws_size,
                              hipStream_t stream) {
    (void)in_sizes; (void)n_in; (void)out_size; (void)ws_size;

    const float* signal = (const float*)d_in[0];

    // workspace: two f16 activation buffers (ping-pong across layers),
    // f16 h ping-pong, barrier counter.  Total ~64.06 MB.
    const long long XE = (long long)BATCH * TLEN * DIM;   // activation elements
    _Float16* xA  = (_Float16*)d_ws;
    _Float16* xB  = xA + XE;
    _Float16* hb  = xB + XE;
    int*      cnt = (int*)(hb + 2 * HS);

    cast_f32_to_f16_kernel<<<(int)((XE + 255) / 256), 256, 0, stream>>>(signal, xA, XE);

    float* out_seq = (float*)d_out;
    float* out_hs  = out_seq + XE;                       // [4,B,512]
    float* out_cs  = out_hs + 4LL * BATCH * DIM;         // [4,B,512]

    for (int l = 0; l < 4; ++l) {
        const float* W = (const float*)d_in[1 + 3 * l];
        const float* R = (const float*)d_in[2 + 3 * l];
        const float* b = (const float*)d_in[3 + 3 * l];
        const _Float16* xin = (l & 1) ? xB : xA;
        _Float16*      xout = (l & 1) ? xA : xB;

        reset_counter_kernel<<<1, 32, 0, stream>>>(cnt);
        lstm_layer_kernel<<<NWG, 256, LDS_TOTAL, stream>>>(
            xin, xout, hb, cnt, W, R, b,
            out_seq,
            out_hs + (size_t)l * BATCH * DIM,
            out_cs + (size_t)l * BATCH * DIM,
            (l == 3) ? 1 : 0);
    }
}
